// MultiTaskGAT_33363305955897
// MI455X (gfx1250) — compile-verified, matrix operand order checked by reading, not run on hardware
//
#include <hip/hip_runtime.h>

typedef __attribute__((ext_vector_type(16))) __bf16 v16bf;
typedef __attribute__((ext_vector_type(8)))  float  v8f;

#define NN    50000
#define NE    500000
#define EDIM  768
#define RDIM  100
#define FDIM  256
#define KIN   2304      // 3*EDIM
#define H1    4
#define HIDD  64
#define C1    256       // H1*HIDD
#define C2    64

static inline int cdiv(int a, int b) { return (a + b - 1) / b; }

// ---------- helpers ----------
__device__ __forceinline__ unsigned f2ord(float f) {
  unsigned u = __float_as_uint(f);
  return (u & 0x80000000u) ? ~u : (u | 0x80000000u);
}
__device__ __forceinline__ float ord2f(unsigned u) {
  return __uint_as_float((u & 0x80000000u) ? (u & 0x7fffffffu) : ~u);
}

// async global->LDS 16B copy (gfx1250 GLOBAL_LOAD_ASYNC_TO_LDS_B128, ASYNCcnt)
__device__ __forceinline__ void async_b128(const void* gaddr, void* lds) {
  unsigned l = (unsigned)(uintptr_t)lds;   // ISA 10.2: LDS aperture uses addr[31:0]
  asm volatile("global_load_async_to_lds_b128 %0, %1, off"
               :: "v"(l), "v"(gaddr) : "memory");
}
__device__ __forceinline__ void wait_async0() {
  asm volatile("s_wait_asynccnt 0" ::: "memory");
}

__device__ __forceinline__ v8f wmma_bf16(v16bf a, v16bf b, v8f c) {
  return __builtin_amdgcn_wmma_f32_16x16x32_bf16(false, a, false, b,
                                                 (short)0, c, false, false);
}

// ---------- prep kernels ----------
// fp32[K][N] -> bf16[N][K]
__global__ void transpose_to_bf16(const float* __restrict__ W, __bf16* __restrict__ Wt,
                                  int K, int N) {
  int i = blockIdx.x * 256 + threadIdx.x;
  if (i >= N * K) return;
  int n = i / K, k = i % K;
  Wt[i] = (__bf16)W[(size_t)k * N + n];
}

// nearest-resize source offsets for the first 2*EDIM concat columns
__global__ void build_colmap(int* __restrict__ colmap) {
  int k = blockIdx.x * 256 + threadIdx.x;
  if (k >= 2 * EDIM) return;
  colmap[k] = (k < EDIM) ? (k * RDIM) / EDIM : ((k - EDIM) * FDIM) / EDIM;
}

// ---------- GEMM1: fused resize/concat gather + bf16 WMMA, bf16 out ----------
// C[50000 x 768] = h0 @ proj_W + proj_b ; h0 built on the fly into LDS
__global__ __launch_bounds__(256)
void gemm1_fused(const float* __restrict__ rand_feat, const float* __restrict__ func_emb,
                 const float* __restrict__ emb, const int* __restrict__ colmap,
                 const __bf16* __restrict__ Bt, const float* __restrict__ bias,
                 __bf16* __restrict__ Cout) {
  __shared__ __bf16 As[2][128 * 64];   // 2 x 16KB
  __shared__ __bf16 Bs[2][64 * 64];    // 2 x 8KB
  const int tid  = threadIdx.x;
  const int lane = tid & 31;
  const int wave = tid >> 5;
  const int n0 = blockIdx.x * 64;
  const int m0 = blockIdx.y * 128;
  const int fsel = (lane & 15) * 64 + ((lane >> 4) << 4);  // fragment row/half select

  v8f acc[4];
  const v8f vz = {0.f, 0.f, 0.f, 0.f, 0.f, 0.f, 0.f, 0.f};
#pragma unroll
  for (int j = 0; j < 4; ++j) acc[j] = vz;

  auto stageB = [&](int t, int buf) {
    int k0 = t * 64;
#pragma unroll
    for (int c = tid; c < 512; c += 256) {           // 512 x 16B = 64x64 bf16
      int nrow = c >> 3, kc = (c & 7) << 3;
      async_b128(Bt + (size_t)(n0 + nrow) * KIN + k0 + kc, &Bs[buf][nrow * 64 + kc]);
    }
  };
  auto stageA = [&](int t, int buf) {
    int k0 = t * 64;
    for (int i = tid; i < 128 * 64; i += 256) {
      int ml = i >> 6, kk = i & 63;
      int m = m0 + ml, k = k0 + kk;
      float v = 0.f;
      if (m < NN) {
        if (k < 2 * EDIM) {
          int cidx = colmap[k];
          v = (k < EDIM) ? rand_feat[(size_t)m * RDIM + cidx]
                         : func_emb[(size_t)m * FDIM + cidx];
        } else {
          v = emb[(size_t)m * EDIM + (k - 2 * EDIM)];
        }
      }
      As[buf][ml * 64 + kk] = (__bf16)v;
    }
  };
  auto compute = [&](int buf) {
    const __bf16* Ab = &As[buf][wave * 16 * 64];
    const __bf16* Bb = &Bs[buf][0];
#pragma unroll
    for (int ks = 0; ks < 2; ++ks) {
      // batch all fragment loads, then burst 4 WMMAs (one DS wait per group)
      v16bf a  = *(const v16bf*)(Ab + ks * 32 + fsel);
      v16bf b0 = *(const v16bf*)(Bb + 0 * 1024 + ks * 32 + fsel);
      v16bf b1 = *(const v16bf*)(Bb + 1 * 1024 + ks * 32 + fsel);
      v16bf b2 = *(const v16bf*)(Bb + 2 * 1024 + ks * 32 + fsel);
      v16bf b3 = *(const v16bf*)(Bb + 3 * 1024 + ks * 32 + fsel);
      acc[0] = wmma_bf16(a, b0, acc[0]);
      acc[1] = wmma_bf16(a, b1, acc[1]);
      acc[2] = wmma_bf16(a, b2, acc[2]);
      acc[3] = wmma_bf16(a, b3, acc[3]);
    }
  };

  const int T = KIN / 64;    // 36
  stageB(0, 0); stageA(0, 0);
  wait_async0();
  __syncthreads();
  for (int t = 0; t < T; ++t) {
    int cur = t & 1;
    if (t + 1 < T) { stageB(t + 1, cur ^ 1); stageA(t + 1, cur ^ 1); }
    compute(cur);
    if (t + 1 < T) { wait_async0(); __syncthreads(); }
  }

  // epilogue: row = m0+wave*16 + r + 8*(lane>>4), col = n0+j*16+(lane&15)
  int rbase = m0 + wave * 16 + ((lane >> 4) << 3);
  int colb  = lane & 15;
#pragma unroll
  for (int j = 0; j < 4; ++j) {
    int col = n0 + j * 16 + colb;
    float bv = bias[col];
#pragma unroll
    for (int r = 0; r < 8; ++r) {
      int row = rbase + r;
      if (row < NN) Cout[(size_t)row * EDIM + col] = (__bf16)(acc[j][r] + bv);
    }
  }
}

// ---------- generic bf16 WMMA GEMM, fp32 out: C[MxN] = A[MxK] @ Bt[NxK]^T (+bias) ----------
__global__ __launch_bounds__(256)
void gemm_bf16_f32(const __bf16* __restrict__ A, const __bf16* __restrict__ Bt,
                   const float* __restrict__ bias, float* __restrict__ C,
                   int M, int N, int K) {
  __shared__ __bf16 Bs[2][64 * 64];    // 2 x 8KB
  const int tid  = threadIdx.x;
  const int lane = tid & 31;
  const int wave = tid >> 5;
  const int n0 = blockIdx.x * 64;
  const int m0 = blockIdx.y * 128;
  const int fsel = (lane & 15) * 64 + ((lane >> 4) << 4);

  v8f acc[4];
  const v8f vz = {0.f, 0.f, 0.f, 0.f, 0.f, 0.f, 0.f, 0.f};
#pragma unroll
  for (int j = 0; j < 4; ++j) acc[j] = vz;

  int arow = m0 + wave * 16 + (lane & 15);
  if (arow >= M) arow = M - 1;   // clamp keeps loads in-bounds; stores are guarded
  const __bf16* Arow = A + (size_t)arow * K + ((lane >> 4) << 4);

  auto stageB = [&](int t, int buf) {
    int k0 = t * 64;
#pragma unroll
    for (int c = tid; c < 512; c += 256) {
      int nrow = c >> 3, kc = (c & 7) << 3;
      async_b128(Bt + (size_t)(n0 + nrow) * K + k0 + kc, &Bs[buf][nrow * 64 + kc]);
    }
  };
  auto compute = [&](int t, int buf) {
    const __bf16* Bb = &Bs[buf][0];
#pragma unroll
    for (int ks = 0; ks < 2; ++ks) {
      v16bf a  = *(const v16bf*)(Arow + t * 64 + ks * 32);
      v16bf b0 = *(const v16bf*)(Bb + 0 * 1024 + ks * 32 + fsel);
      v16bf b1 = *(const v16bf*)(Bb + 1 * 1024 + ks * 32 + fsel);
      v16bf b2 = *(const v16bf*)(Bb + 2 * 1024 + ks * 32 + fsel);
      v16bf b3 = *(const v16bf*)(Bb + 3 * 1024 + ks * 32 + fsel);
      acc[0] = wmma_bf16(a, b0, acc[0]);
      acc[1] = wmma_bf16(a, b1, acc[1]);
      acc[2] = wmma_bf16(a, b2, acc[2]);
      acc[3] = wmma_bf16(a, b3, acc[3]);
    }
  };

  const int T = K / 64;
  stageB(0, 0);
  wait_async0();
  __syncthreads();
  for (int t = 0; t < T; ++t) {
    int cur = t & 1;
    if (t + 1 < T) stageB(t + 1, cur ^ 1);
    compute(t, cur);
    if (t + 1 < T) { wait_async0(); __syncthreads(); }
  }

  int rbase = m0 + wave * 16 + ((lane >> 4) << 3);
  int colb  = lane & 15;
#pragma unroll
  for (int j = 0; j < 4; ++j) {
    int col = n0 + j * 16 + colb;
    float bv = bias ? bias[col] : 0.f;
#pragma unroll
    for (int r = 0; r < 8; ++r) {
      int row = rbase + r;
      if (row < M) C[(size_t)row * N + col] = acc[j][r] + bv;
    }
  }
}

// ---------- attention ----------
__global__ void attn_lr(const float* __restrict__ z, const float* __restrict__ al,
                        const float* __restrict__ ar, float* __restrict__ el,
                        float* __restrict__ er, int H, int D, int total) {
  int i = blockIdx.x * 256 + threadIdx.x;   // over NN*H
  if (i >= total) return;
  int h = i % H;
  const float* zr = z + (size_t)i * D;
  float sl = 0.f, sr = 0.f;
  for (int d = 0; d < D; ++d) { float v = zr[d]; sl += v * al[h * D + d]; sr += v * ar[h * D + d]; }
  el[i] = sl; er[i] = sr;
}

__global__ void init_md(unsigned* __restrict__ mo, float* __restrict__ denom, int count) {
  int i = blockIdx.x * 256 + threadIdx.x;
  if (i < count) { mo[i] = 0u; denom[i] = 0.f; }
}

__global__ void init_out_bias(float* __restrict__ out, const float* __restrict__ b,
                              int total, int C) {
  int i = blockIdx.x * 256 + threadIdx.x;
  if (i < total) out[i] = b[i % C];
}

__global__ void edge_max(const int* __restrict__ src, const int* __restrict__ dst,
                         const float* __restrict__ el, const float* __restrict__ er,
                         float* __restrict__ ebuf, unsigned* __restrict__ mo,
                         int H, int total) {
  int i = blockIdx.x * 256 + threadIdx.x;   // over NE*H
  if (i >= total) return;
  int e = i / H, h = i % H;
  int s = src[e], d = dst[e];
  float x = el[s * H + h] + er[d * H + h];
  x = x > 0.f ? x : 0.2f * x;               // leaky_relu(0.2)
  ebuf[i] = x;
  atomicMax(&mo[d * H + h], f2ord(x));
}

__global__ void edge_exp(const int* __restrict__ dst, const unsigned* __restrict__ mo,
                         float* __restrict__ ebuf, float* __restrict__ denom,
                         int H, int total) {
  int i = blockIdx.x * 256 + threadIdx.x;
  if (i >= total) return;
  int e = i / H, h = i % H;
  int d = dst[e];
  float ex = __expf(ebuf[i] - ord2f(mo[d * H + h]));
  ebuf[i] = ex;
  atomicAdd(&denom[d * H + h], ex);
}

__global__ void edge_aggr(const int* __restrict__ src, const int* __restrict__ dst,
                          const float* __restrict__ ebuf, const float* __restrict__ denom,
                          const float* __restrict__ z, float* __restrict__ out,
                          int H, int D, int total) {
  int i = blockIdx.x * 256 + threadIdx.x;
  if (i >= total) return;
  int e = i / H, h = i % H;
  int s = src[e], d = dst[e];
  float alpha = ebuf[i] / denom[d * H + h];
  const float* zs = z + ((size_t)s * H + h) * D;
  float* od = out + ((size_t)d * H + h) * D;
  for (int k = 0; k < D; k += 4) {
    float4 v = *(const float4*)(zs + k);
    atomicAdd(od + k + 0, alpha * v.x);
    atomicAdd(od + k + 1, alpha * v.y);
    atomicAdd(od + k + 2, alpha * v.z);
    atomicAdd(od + k + 3, alpha * v.w);
  }
}

__global__ void f32_to_bf16_k(const float* __restrict__ in, __bf16* __restrict__ out, int total) {
  int i = blockIdx.x * 256 + threadIdx.x;
  if (i < total) out[i] = (__bf16)in[i];
}

// ---------- readout ----------
__global__ void col_mean(const float* __restrict__ h, float* __restrict__ hg) {
  __shared__ float red[256];
  int c = blockIdx.x;                         // 0..63
  float s = 0.f;
  for (int n = threadIdx.x; n < NN; n += 256) s += h[(size_t)n * C2 + c];
  red[threadIdx.x] = s; __syncthreads();
  for (int st = 128; st > 0; st >>= 1) {
    if (threadIdx.x < st) red[threadIdx.x] += red[threadIdx.x + st];
    __syncthreads();
  }
  if (threadIdx.x == 0) hg[c] = red[0] / (float)NN;
}

__global__ void node_mlp(const float* __restrict__ h, const float* __restrict__ W1,
                         const float* __restrict__ b1, const float* __restrict__ W2,
                         const float* __restrict__ b2, float* __restrict__ out) {
  int n = blockIdx.x * 256 + threadIdx.x;
  if (n >= NN) return;
  float hr[64];
  for (int k = 0; k < 64; ++k) hr[k] = h[(size_t)n * 64 + k];
  float l0 = b2[0], l1 = b2[1];
  for (int j = 0; j < 64; ++j) {
    float a = b1[j];
    for (int k = 0; k < 64; ++k) a += hr[k] * W1[k * 64 + j];
    a = a > 0.f ? a : 0.f;
    l0 += a * W2[j * 2 + 0];
    l1 += a * W2[j * 2 + 1];
  }
  out[n * 2 + 0] = l0;
  out[n * 2 + 1] = l1;
}

__global__ void graph_mlp(const float* __restrict__ hg, const float* __restrict__ W1,
                          const float* __restrict__ b1, const float* __restrict__ W2,
                          const float* __restrict__ b2, float* __restrict__ out2) {
  __shared__ float t[64];
  int j = threadIdx.x;
  if (j < 64) {
    float a = b1[j];
    for (int k = 0; k < 64; ++k) a += hg[k] * W1[k * 64 + j];
    t[j] = a > 0.f ? a : 0.f;
  }
  __syncthreads();
  if (j < 2) {
    float l = b2[j];
    for (int k = 0; k < 64; ++k) l += t[k] * W2[k * 2 + j];
    out2[j] = l;
  }
}

// ---------- host ----------
extern "C" void kernel_launch(void* const* d_in, const int* in_sizes, int n_in,
                              void* d_out, int out_size, void* d_ws, size_t ws_size,
                              hipStream_t stream) {
  const float* rand_feat = (const float*)d_in[0];
  const float* func_emb  = (const float*)d_in[1];
  const float* emb       = (const float*)d_in[2];
  const int*   src       = (const int*)d_in[3];
  const int*   dst       = (const int*)d_in[4];
  const float* proj_W    = (const float*)d_in[5];
  const float* proj_b    = (const float*)d_in[6];
  const float* g1_W      = (const float*)d_in[7];
  const float* g1_al     = (const float*)d_in[8];
  const float* g1_ar     = (const float*)d_in[9];
  const float* g1_b      = (const float*)d_in[10];
  const float* g2_W      = (const float*)d_in[11];
  const float* g2_al     = (const float*)d_in[12];
  const float* g2_ar     = (const float*)d_in[13];
  const float* g2_b      = (const float*)d_in[14];
  const float* n_W1      = (const float*)d_in[15];
  const float* n_b1      = (const float*)d_in[16];
  const float* n_W2      = (const float*)d_in[17];
  const float* n_b2      = (const float*)d_in[18];
  const float* gm_W1     = (const float*)d_in[19];
  const float* gm_b1     = (const float*)d_in[20];
  const float* gm_W2     = (const float*)d_in[21];
  const float* gm_b2     = (const float*)d_in[22];

  char* w = (char*)d_ws;
  size_t off = 0;
  auto alloc = [&](size_t bytes) -> void* {
    void* p = w + off;
    off = (off + bytes + 255) & ~(size_t)255;
    return p;
  };
  __bf16*   h1bf  = (__bf16*)alloc((size_t)NN * EDIM * 2);   // A: h1 bf16 / later h2 bf16
  float*    zbuf  = (float*) alloc((size_t)NN * C1 * 4);     // B: z1 / later z2
  float*    obuf  = (float*) alloc((size_t)NN * C1 * 4);     // C: out1 / later out2 (final h)
  float*    el    = (float*) alloc((size_t)NN * H1 * 4);
  float*    er    = (float*) alloc((size_t)NN * H1 * 4);
  unsigned* mo    = (unsigned*)alloc((size_t)NN * H1 * 4);
  float*    denom = (float*) alloc((size_t)NN * H1 * 4);
  float*    ebuf  = (float*) alloc((size_t)NE * H1 * 4);
  __bf16*   projWt= (__bf16*)alloc((size_t)EDIM * KIN * 2);
  __bf16*   g1Wt  = (__bf16*)alloc((size_t)C1 * EDIM * 2);
  __bf16*   g2Wt  = (__bf16*)alloc((size_t)C2 * C1 * 2);
  int*      colmap= (int*)   alloc((size_t)2 * EDIM * 4);
  float*    hg    = (float*) alloc(64 * 4);
  if (off > ws_size) return;   // insufficient workspace

  // 1) weight prep: bf16 transposed [N][K] + gather map
  transpose_to_bf16<<<cdiv(EDIM * KIN, 256), 256, 0, stream>>>(proj_W, projWt, KIN, EDIM);
  transpose_to_bf16<<<cdiv(C1 * EDIM, 256), 256, 0, stream>>>(g1_W, g1Wt, EDIM, C1);
  transpose_to_bf16<<<cdiv(C2 * C1, 256), 256, 0, stream>>>(g2_W, g2Wt, C1, C2);
  build_colmap<<<cdiv(2 * EDIM, 256), 256, 0, stream>>>(colmap);

  // 2) fused resize/concat + projection GEMM (bf16 out)
  dim3 grd1(EDIM / 64, cdiv(NN, 128));
  gemm1_fused<<<grd1, 256, 0, stream>>>(rand_feat, func_emb, emb, colmap, projWt, proj_b, h1bf);

  // 3) z1 = h1 @ g1_W  [NN x 256]
  dim3 grd2(C1 / 64, cdiv(NN, 128));
  gemm_bf16_f32<<<grd2, 256, 0, stream>>>(h1bf, g1Wt, nullptr, zbuf, NN, C1, EDIM);

  // 4) GAT layer 1 edge softmax + aggregation
  attn_lr<<<cdiv(NN * H1, 256), 256, 0, stream>>>(zbuf, g1_al, g1_ar, el, er, H1, HIDD, NN * H1);
  init_md<<<cdiv(NN * H1, 256), 256, 0, stream>>>(mo, denom, NN * H1);
  init_out_bias<<<cdiv(NN * C1, 256), 256, 0, stream>>>(obuf, g1_b, NN * C1, C1);
  edge_max<<<cdiv(NE * H1, 256), 256, 0, stream>>>(src, dst, el, er, ebuf, mo, H1, NE * H1);
  edge_exp<<<cdiv(NE * H1, 256), 256, 0, stream>>>(dst, mo, ebuf, denom, H1, NE * H1);
  edge_aggr<<<cdiv(NE * H1, 256), 256, 0, stream>>>(src, dst, ebuf, denom, zbuf, obuf, H1, HIDD, NE * H1);

  // 5) h2 = bf16(out1)  (reuse region A)
  f32_to_bf16_k<<<cdiv(NN * C1, 256), 256, 0, stream>>>(obuf, h1bf, NN * C1);

  // 6) z2 = h2 @ g2_W  [NN x 64] (reuse region B)
  dim3 grd3(C2 / 64, cdiv(NN, 128));
  gemm_bf16_f32<<<grd3, 256, 0, stream>>>(h1bf, g2Wt, nullptr, zbuf, NN, C2, C1);

  // 7) GAT layer 2 (H=1) -> final h in obuf (reuse region C)
  attn_lr<<<cdiv(NN, 256), 256, 0, stream>>>(zbuf, g2_al, g2_ar, el, er, 1, HIDD, NN);
  init_md<<<cdiv(NN, 256), 256, 0, stream>>>(mo, denom, NN);
  init_out_bias<<<cdiv(NN * C2, 256), 256, 0, stream>>>(obuf, g2_b, NN * C2, C2);
  edge_max<<<cdiv(NE, 256), 256, 0, stream>>>(src, dst, el, er, ebuf, mo, 1, NE);
  edge_exp<<<cdiv(NE, 256), 256, 0, stream>>>(dst, mo, ebuf, denom, 1, NE);
  edge_aggr<<<cdiv(NE, 256), 256, 0, stream>>>(src, dst, ebuf, denom, zbuf, obuf, 1, HIDD, NE);

  // 8) readout: node logits + graph logits
  node_mlp<<<cdiv(NN, 256), 256, 0, stream>>>(obuf, n_W1, n_b1, n_W2, n_b2, (float*)d_out);
  col_mean<<<64, 256, 0, stream>>>(obuf, hg);
  graph_mlp<<<1, 64, 0, stream>>>(hg, gm_W1, gm_b1, gm_W2, gm_b2, (float*)d_out + (size_t)NN * 2);
}
